// GPTLanguageModel_48713519071497
// MI455X (gfx1250) — matrix operand order repaired and checked by reading, not compiled, not run on hardware
//
#include <hip/hip_runtime.h>
#include <hip/hip_bf16.h>
#include <math.h>
#include <stdint.h>

#define Bsz 16
#define Tt  512
#define Ee  512
#define NH  8
#define Ll  4
#define Vv  2048
#define HSz 64
#define FFd 2048

typedef __bf16 bf16;
typedef __attribute__((ext_vector_type(16))) __bf16 v16bf;
typedef __attribute__((ext_vector_type(8)))  float  v8f;

// ---------------------------------------------------------------------------
// WMMA helpers (CDNA5 v_wmma_f32_16x16x32_bf16, wave32)
// ---------------------------------------------------------------------------
__device__ __forceinline__ v8f wmma_bf16(v16bf a, v16bf b, v8f c) {
  return __builtin_amdgcn_wmma_f32_16x16x32_bf16(
      /*neg_a=*/false, a, /*neg_b=*/false, b,
      /*c_mod=*/(short)0, c, /*reuse_a=*/false, /*reuse_b=*/false);
}

// A fragment: rows m0..m0+15, K k0..k0+31, src row-major (leading dim ld).
// Lane layout: lanes 0-15 hold K 0-15, lanes 16-31 hold K 16-31.
__device__ __forceinline__ v16bf load_afrag(const float* S, int m0, int ld, int k0) {
  const int l = threadIdx.x & 31;
  const float* row = S + (size_t)(m0 + (l & 15)) * ld + k0 + ((l >= 16) ? 16 : 0);
  v16bf f;
#pragma unroll
  for (int v = 0; v < 4; ++v) {
    f[2 * v]     = (bf16)row[2 * v];
    f[2 * v + 1] = (bf16)row[2 * v + 1];
  }
#pragma unroll
  for (int v = 0; v < 4; ++v) {
    f[8 + 2 * v] = (bf16)row[8 + 2 * v];
    f[9 + 2 * v] = (bf16)row[8 + 2 * v + 1];
  }
  return f;
}

// B fragment: K rows k0..k0+31, cols n0..n0+15, src row-major [k][n] leading dim ld.
// Lanes 0-15 hold K 0-15, lanes 16-31 hold K 16-31.
__device__ __forceinline__ v16bf load_bfrag(const float* Sb, int k0, int ld, int n0) {
  const int l = threadIdx.x & 31;
  const float* col = Sb + (size_t)(k0 + ((l >= 16) ? 16 : 0)) * ld + n0 + (l & 15);
  v16bf f;
#pragma unroll
  for (int v = 0; v < 8; ++v) {
    f[2 * v]     = (bf16)col[(2 * v) * ld];
    f[2 * v + 1] = (bf16)col[(2 * v + 1) * ld];
  }
  return f;
}

// B fragment from an [n][k] (k contiguous) source, e.g. K^T with K stored (T,HS).
__device__ __forceinline__ v16bf load_bfrag_t(const float* Sb, int k0, int ldn, int n0) {
  const int l = threadIdx.x & 31;
  const float* col = Sb + (size_t)(n0 + (l & 15)) * ldn + k0 + ((l >= 16) ? 16 : 0);
  v16bf f;
#pragma unroll
  for (int v = 0; v < 8; ++v) {
    f[2 * v]     = (bf16)col[2 * v];
    f[2 * v + 1] = (bf16)col[2 * v + 1];
  }
  return f;
}

// ---------------------------------------------------------------------------
// Async-stage one 128x32 A tile + 32x64 B tile (fp32) into LDS: 6 b128 ops
// per thread (256 threads), tracked by ASYNCcnt.
// ---------------------------------------------------------------------------
__device__ __forceinline__ void issue_tile_async(
    const float* __restrict__ A, const float* __restrict__ Bm,
    int K, int N, int gm0, int gn0, int kt,
    unsigned asBase, unsigned bsBase)
{
  const int tid = threadIdx.x;
#pragma unroll
  for (int i = 0; i < 4; ++i) {
    int f = tid + 256 * i;            // float4 id 0..1023 (128x32 tile)
    int ar = f >> 3, ac4 = f & 7;     // 8 float4 per 32-col row
    const float* gA = A + (size_t)(gm0 + ar) * K + kt + ac4 * 4;
    unsigned la = asBase + (unsigned)((ar * 32 + ac4 * 4) * 4);
    asm volatile("global_load_async_to_lds_b128 %0, %1, off"
                 :: "v"(la), "v"(gA) : "memory");
  }
#pragma unroll
  for (int i = 0; i < 2; ++i) {
    int f = tid + 256 * i;            // float4 id 0..511 (32x64 tile)
    int br = f >> 4, bc4 = f & 15;    // 16 float4 per 64-col row
    const float* gB = Bm + (size_t)(kt + br) * N + gn0 + bc4 * 4;
    unsigned lb = bsBase + (unsigned)((br * 64 + bc4 * 4) * 4);
    asm volatile("global_load_async_to_lds_b128 %0, %1, off"
                 :: "v"(lb), "v"(gB) : "memory");
  }
}

// ---------------------------------------------------------------------------
// GEMM: C[M,N] = epilogue(A[M,K] @ B[K,N]); fp32 row-major, bf16 WMMA math.
// 256 threads (8 waves), 128x64 block tile, BK=32, double-buffered async
// global->LDS staging so tile k+1 DMA overlaps tile k WMMA.
// ---------------------------------------------------------------------------
__global__ __launch_bounds__(256) void gemm_kernel(
    const float* __restrict__ A, const float* __restrict__ Bm,
    float* __restrict__ C, const float* __restrict__ bias,
    const float* __restrict__ resid, int M, int N, int K, int relu)
{
  __shared__ float As[2][128 * 32];   // 2 x 16 KB
  __shared__ float Bs[2][32 * 64];    // 2 x  8 KB
  const int gm0 = blockIdx.y * 128;
  const int gn0 = blockIdx.x * 64;
  const int tid = threadIdx.x;
  const int w   = tid >> 5;
  const int wm0 = (w >> 1) * 32;      // 4 wave-rows
  const int wn0 = (w & 1) * 32;       // 2 wave-cols
  v8f acc[4] = {};

  const unsigned asB0 = (unsigned)(size_t)(&As[0][0]);
  const unsigned asB1 = (unsigned)(size_t)(&As[1][0]);
  const unsigned bsB0 = (unsigned)(size_t)(&Bs[0][0]);
  const unsigned bsB1 = (unsigned)(size_t)(&Bs[1][0]);

  int cur = 0;
  issue_tile_async(A, Bm, K, N, gm0, gn0, 0, asB0, bsB0);

  for (int kt = 0; kt < K; kt += 32) {
    if (kt + 32 < K) {
      // prefetch next tile into the other buffer, then wait for current:
      // in-order ASYNCcnt completion => "<= 6 outstanding" means only the
      // 6 just-issued ops may still be in flight.
      issue_tile_async(A, Bm, K, N, gm0, gn0, kt + 32,
                       cur ? asB0 : asB1, cur ? bsB0 : bsB1);
      asm volatile("s_wait_asynccnt 0x6" ::: "memory");
    } else {
      asm volatile("s_wait_asynccnt 0x0" ::: "memory");
    }
    __syncthreads();

    const float* Ab = &As[cur][0];
    const float* Bb = &Bs[cur][0];
    v16bf a0 = load_afrag(Ab, wm0,      32, 0);
    v16bf a1 = load_afrag(Ab, wm0 + 16, 32, 0);
    v16bf b0 = load_bfrag(Bb, 0, 64, wn0);
    v16bf b1 = load_bfrag(Bb, 0, 64, wn0 + 16);
    acc[0] = wmma_bf16(a0, b0, acc[0]);
    acc[1] = wmma_bf16(a0, b1, acc[1]);
    acc[2] = wmma_bf16(a1, b0, acc[2]);
    acc[3] = wmma_bf16(a1, b1, acc[3]);
    __syncthreads();   // all waves done reading buf[cur] before it is refilled
    cur ^= 1;
  }

  const int l    = tid & 31;
  const int moff = (l >= 16) ? 8 : 0;
#pragma unroll
  for (int t = 0; t < 4; ++t) {
    const int mb = gm0 + wm0 + ((t >> 1) ? 16 : 0);
    const int n  = gn0 + wn0 + ((t & 1) ? 16 : 0) + (l & 15);
#pragma unroll
    for (int v = 0; v < 8; ++v) {
      const int m = mb + moff + v;
      float val = acc[t][v];
      if (bias)  val += bias[n];
      if (relu)  val = fmaxf(val, 0.0f);
      if (resid) val += resid[(size_t)m * N + n];
      C[(size_t)m * N + n] = val;
    }
  }
}

// ---------------------------------------------------------------------------
// Attention (per block: one (b,h) and 16 query rows): scores = q@k^T*scale +
// rel term, causal softmax, ctx = att@v, written in (B,T,E) layout.
// Heads are contiguous (T,HS) blocks thanks to the reference's view semantics.
// ---------------------------------------------------------------------------
__global__ __launch_bounds__(128) void attn_kernel(
    const float* __restrict__ q, const float* __restrict__ k,
    const float* __restrict__ v, const float* __restrict__ rel,
    float* __restrict__ out)
{
  __shared__ float Qs[16 * 64];        // 4 KB
  __shared__ float S[16 * 512];        // 32 KB score/prob tile
  __shared__ float red[16 * 8];
  __shared__ float rowstat[16];

  const int nqb = Tt / 16;             // 32
  const int bh  = blockIdx.x / nqb;
  const int qb  = blockIdx.x % nqb;
  const int b   = bh / NH;
  const int h   = bh % NH;
  const int qi0 = qb * 16;

  const float* qp = q + (size_t)b * Tt * Ee + (size_t)h * Tt * HSz;
  const float* kp = k + (size_t)b * Tt * Ee + (size_t)h * Tt * HSz;
  const float* vp = v + (size_t)b * Tt * Ee + (size_t)h * Tt * HSz;

  const int tid = threadIdx.x;
  // Stage Q tile 16x64
  for (int i = tid; i < 16 * 64 / 4; i += 128) {
    int r = i >> 4, c4 = i & 15;
    ((float4*)Qs)[r * 16 + c4] =
        *(const float4*)(qp + (size_t)(qi0 + r) * HSz + c4 * 4);
  }
  __syncthreads();

  const int w    = tid >> 5;
  const int l    = tid & 31;
  const int moff = (l >= 16) ? 8 : 0;
  const float scale = 0.125f;          // 1/sqrt(64)

  // scores: each wave covers 128 key columns (8 WMMA tiles), 2 K-steps of 32
  v16bf aq0 = load_afrag(Qs, 0, 64, 0);
  v16bf aq1 = load_afrag(Qs, 0, 64, 32);
  const int c0 = w * 128;
  for (int ct = 0; ct < 8; ++ct) {
    const int j0 = c0 + ct * 16;
    v8f acc = {};
    acc = wmma_bf16(aq0, load_bfrag_t(kp, 0,  HSz, j0), acc);
    acc = wmma_bf16(aq1, load_bfrag_t(kp, 32, HSz, j0), acc);
    const int jn = j0 + (l & 15);
#pragma unroll
    for (int vv = 0; vv < 8; ++vv) S[(moff + vv) * 512 + jn] = acc[vv] * scale;
  }
  __syncthreads();

  // relative-position term + causal mask (VALU dot: depends on i-j, not a GEMM)
  for (int e = tid; e < 16 * 512; e += 128) {
    const int i = e >> 9, j = e & 511;
    const int gi = qi0 + i;
    float s;
    if (j > gi) {
      s = -INFINITY;
    } else {
      const float* rr = rel + (size_t)(gi - j + (Tt - 1)) * HSz;
      const float* qq = Qs + i * 64;
      float dot = 0.0f;
#pragma unroll
      for (int d = 0; d < 64; d += 4)
        dot += qq[d] * rr[d] + qq[d + 1] * rr[d + 1] +
               qq[d + 2] * rr[d + 2] + qq[d + 3] * rr[d + 3];
      s = S[i * 512 + j] + dot;
    }
    S[i * 512 + j] = s;
  }
  __syncthreads();

  // row softmax: 8 threads per row, 64 cols each
  {
    const int row = tid >> 3, part = tid & 7;
    float* Sr = S + row * 512 + part * 64;
    float mx = -INFINITY;
    for (int j = 0; j < 64; ++j) mx = fmaxf(mx, Sr[j]);
    red[row * 8 + part] = mx;
    __syncthreads();
    if (part == 0) {
      float m = red[row * 8];
      for (int i = 1; i < 8; ++i) m = fmaxf(m, red[row * 8 + i]);
      rowstat[row] = m;
    }
    __syncthreads();
    const float m = rowstat[row];
    float sum = 0.0f;
    for (int j = 0; j < 64; ++j) {
      float ev = __expf(Sr[j] - m);
      Sr[j] = ev;
      sum += ev;
    }
    red[row * 8 + part] = sum;
    __syncthreads();
    if (part == 0) {
      float t = 0.0f;
      for (int i = 0; i < 8; ++i) t += red[row * 8 + i];
      rowstat[row] = 1.0f / t;
    }
    __syncthreads();
    const float inv = rowstat[row];
    for (int j = 0; j < 64; ++j) Sr[j] *= inv;
  }
  __syncthreads();

  // ctx = att(16x512) @ V(512x64); wave w handles 16 output cols d0 = w*16
  float* outp = out + (size_t)b * Tt * Ee + (size_t)h * HSz;
  {
    const int d0 = w * 16;
    v8f acc = {};
    for (int j0 = 0; j0 < 512; j0 += 32) {
      v16bf ap = load_afrag(S, 0, 512, j0);
      v16bf bv = load_bfrag(vp, j0, HSz, d0);
      acc = wmma_bf16(ap, bv, acc);
    }
    const int n = d0 + (l & 15);
#pragma unroll
    for (int vv = 0; vv < 8; ++vv)
      outp[(size_t)(qi0 + moff + vv) * Ee + n] = acc[vv];
  }
}

// ---------------------------------------------------------------------------
// LayerNorm over E=512, one block (256 threads) per row
// ---------------------------------------------------------------------------
__global__ __launch_bounds__(256) void ln_kernel(
    const float* __restrict__ x, const float* __restrict__ g,
    const float* __restrict__ bb, float* __restrict__ o)
{
  __shared__ float warpsum[8];
  __shared__ float stat[2];
  const int row = blockIdx.x;
  const float* xr = x + (size_t)row * Ee;
  float* orow = o + (size_t)row * Ee;
  const int tid = threadIdx.x;
  float v0 = xr[tid], v1 = xr[tid + 256];
  float s = v0 + v1;
  for (int off = 16; off; off >>= 1) s += __shfl_down(s, off, 32);
  if ((tid & 31) == 0) warpsum[tid >> 5] = s;
  __syncthreads();
  if (tid == 0) {
    float t = 0.0f;
    for (int i = 0; i < 8; ++i) t += warpsum[i];
    stat[0] = t * (1.0f / Ee);
  }
  __syncthreads();
  const float m = stat[0];
  const float d0 = v0 - m, d1 = v1 - m;
  float s2 = d0 * d0 + d1 * d1;
  for (int off = 16; off; off >>= 1) s2 += __shfl_down(s2, off, 32);
  if ((tid & 31) == 0) warpsum[tid >> 5] = s2;
  __syncthreads();
  if (tid == 0) {
    float t = 0.0f;
    for (int i = 0; i < 8; ++i) t += warpsum[i];
    stat[1] = rsqrtf(t * (1.0f / Ee) + 1e-5f);
  }
  __syncthreads();
  const float r = stat[1];
  orow[tid]       = d0 * r * g[tid]       + bb[tid];
  orow[tid + 256] = d1 * r * g[tid + 256] + bb[tid + 256];
}

// ---------------------------------------------------------------------------
// Embedding: x[b,t,:] = tok_emb[idx[b,t],:] + pos_emb[t,:]
// ---------------------------------------------------------------------------
__global__ __launch_bounds__(128) void embed_kernel(
    const int* __restrict__ idx, const float* __restrict__ tok,
    const float* __restrict__ pos, float* __restrict__ x)
{
  const int bt = blockIdx.x;
  const int t = bt % Tt;
  const int token = idx[bt];
  const float* te = tok + (size_t)token * Ee;
  const float* pe = pos + (size_t)t * Ee;
  float* xr = x + (size_t)bt * Ee;
  for (int i = threadIdx.x; i < Ee; i += 128) xr[i] = te[i] + pe[i];
}

// ---------------------------------------------------------------------------
extern "C" void kernel_launch(void* const* d_in, const int* in_sizes, int n_in,
                              void* d_out, int out_size, void* d_ws, size_t ws_size,
                              hipStream_t stream)
{
  const int*   idx  = (const int*)d_in[0];
  const float* tok  = (const float*)d_in[1];
  const float* pos  = (const float*)d_in[2];
  const float* Wq   = (const float*)d_in[3];
  const float* Wk   = (const float*)d_in[4];
  const float* Wv   = (const float*)d_in[5];
  const float* rel  = (const float*)d_in[6];
  const float* Wo   = (const float*)d_in[7];
  const float* bo   = (const float*)d_in[8];
  const float* ln1g = (const float*)d_in[9];
  const float* ln1b = (const float*)d_in[10];
  const float* W1   = (const float*)d_in[11];
  const float* b1   = (const float*)d_in[12];
  const float* W2   = (const float*)d_in[13];
  const float* b2   = (const float*)d_in[14];
  const float* ln2g = (const float*)d_in[15];
  const float* ln2b = (const float*)d_in[16];
  const float* lnfg = (const float*)d_in[17];
  const float* lnfb = (const float*)d_in[18];
  const float* Wlm  = (const float*)d_in[19];
  const float* blm  = (const float*)d_in[20];
  float* out = (float*)d_out;

  const size_t NTOK = (size_t)Bsz * Tt;   // 8192
  float* ws  = (float*)d_ws;
  float* x   = ws;
  float* hbf = x   + NTOK * Ee;
  float* qb  = hbf + NTOK * Ee;
  float* kb  = qb  + NTOK * Ee;
  float* vb  = kb  + NTOK * Ee;
  float* ctx = vb  + NTOK * Ee;
  float* ff  = ctx + NTOK * Ee;           // NTOK * FF

  embed_kernel<<<(int)NTOK, 128, 0, stream>>>(idx, tok, pos, x);

  const dim3 gE(Ee / 64, (int)NTOK / 128);
  const dim3 gF(FFd / 64, (int)NTOK / 128);
  const dim3 gV(Vv / 64, (int)NTOK / 128);

  for (int l = 0; l < Ll; ++l) {
    const float* wq = Wq + (size_t)l * Ee * Ee;
    const float* wk = Wk + (size_t)l * Ee * Ee;
    const float* wv = Wv + (size_t)l * Ee * Ee;
    const float* wo = Wo + (size_t)l * Ee * Ee;
    const float* rl = rel + (size_t)l * (2 * Tt - 1) * HSz;

    ln_kernel<<<(int)NTOK, 256, 0, stream>>>(x, ln1g + l * Ee, ln1b + l * Ee, hbf);
    gemm_kernel<<<gE, 256, 0, stream>>>(hbf, wq, qb, nullptr, nullptr,
                                        (int)NTOK, Ee, Ee, 0);
    gemm_kernel<<<gE, 256, 0, stream>>>(hbf, wk, kb, nullptr, nullptr,
                                        (int)NTOK, Ee, Ee, 0);
    gemm_kernel<<<gE, 256, 0, stream>>>(hbf, wv, vb, nullptr, nullptr,
                                        (int)NTOK, Ee, Ee, 0);
    attn_kernel<<<Bsz * NH * (Tt / 16), 128, 0, stream>>>(qb, kb, vb, rl, ctx);
    gemm_kernel<<<gE, 256, 0, stream>>>(ctx, wo, x, bo + l * Ee, x,
                                        (int)NTOK, Ee, Ee, 0);
    ln_kernel<<<(int)NTOK, 256, 0, stream>>>(x, ln2g + l * Ee, ln2b + l * Ee, hbf);
    gemm_kernel<<<gF, 256, 0, stream>>>(hbf, W1 + (size_t)l * Ee * FFd, ff,
                                        b1 + l * FFd, nullptr,
                                        (int)NTOK, FFd, Ee, 1);
    gemm_kernel<<<gE, 256, 0, stream>>>(ff, W2 + (size_t)l * FFd * Ee, x,
                                        b2 + l * Ee, x,
                                        (int)NTOK, Ee, FFd, 0);
  }
  ln_kernel<<<(int)NTOK, 256, 0, stream>>>(x, lnfg, lnfb, hbf);
  gemm_kernel<<<gV, 256, 0, stream>>>(hbf, Wlm, out, blm, nullptr,
                                      (int)NTOK, Vv, Ee, 0);
}
